// Arcface_7816840479108
// MI455X (gfx1250) — matrix operand order repaired and checked by reading, not compiled, not run on hardware
//
#include <hip/hip_runtime.h>
#include <hip/hip_bf16.h>

// ---------------- types ----------------
typedef __attribute__((ext_vector_type(16))) _Float16 v16h;
typedef __attribute__((ext_vector_type(8)))  _Float16 v8h;
typedef __attribute__((ext_vector_type(4)))  _Float16 h4;
typedef __attribute__((ext_vector_type(8)))  float    v8f;
typedef __attribute__((ext_vector_type(16))) float    v16f;
typedef __attribute__((ext_vector_type(4)))  float    f4;

#define B_  32
#define S_  16
#define W_  128
#define H_  256
#define P_  120                     // S*(S-1)/2
#define NPART 1024                  // partial-sum blocks for norm reduction

// ---------------- kernel 1: per-block partial sum of squares ----------------
__global__ __launch_bounds__(256) void arcface_sumsq_partial(
    const float* __restrict__ x, float* __restrict__ partials, int n4) {
  __shared__ float red[256];
  const f4* x4 = (const f4*)x;
  float s = 0.f;
  for (int idx = blockIdx.x * blockDim.x + threadIdx.x; idx < n4;
       idx += gridDim.x * blockDim.x) {
    f4 v = x4[idx];
    s += v.x * v.x + v.y * v.y + v.z * v.z + v.w * v.w;
  }
  red[threadIdx.x] = s;
  __syncthreads();
  for (int off = 128; off > 0; off >>= 1) {
    if (threadIdx.x < off) red[threadIdx.x] += red[threadIdx.x + off];
    __syncthreads();
  }
  if (threadIdx.x == 0) partials[blockIdx.x] = red[0];
}

// ---------------- kernel 2: deterministic final reduce -> norm^2 ----------------
__global__ __launch_bounds__(256) void arcface_sumsq_final(
    const float* __restrict__ partials, float* __restrict__ normbuf, int n) {
  __shared__ float red[256];
  float s = 0.f;
  for (int i = threadIdx.x; i < n; i += 256) s += partials[i];
  red[threadIdx.x] = s;
  __syncthreads();
  for (int off = 128; off > 0; off >>= 1) {
    if (threadIdx.x < off) red[threadIdx.x] += red[threadIdx.x + off];
    __syncthreads();
  }
  if (threadIdx.x == 0) normbuf[0] = red[0];
}

// helper: pack a0..a3 in order (two 8-float k runs, 16 apart) into a WMMA-ordered v16h
static __device__ __forceinline__ v16h cvt16(f4 a0, f4 a1, f4 a2, f4 a3) {
  v8f lo = __builtin_shufflevector(a0, a1, 0, 1, 2, 3, 4, 5, 6, 7);
  v8f hi = __builtin_shufflevector(a2, a3, 0, 1, 2, 3, 4, 5, 6, 7);
  v16f f = __builtin_shufflevector(lo, hi, 0, 1, 2, 3, 4, 5, 6, 7, 8, 9, 10,
                                   11, 12, 13, 14, 15);
  return __builtin_convertvector(f, v16h);
}

// ---------------- kernel 3: batched 128x128x256 GEMM via WMMA + fused argmax ----
// grid = (P_, B_), block = 256 threads (8 waves). One (b,p) pair per workgroup.
// Dynamic LDS: 64 KB f16 B-operand in fragment layout + 128 ints of t1.
__global__ __launch_bounds__(256) void arcface_scores_kernel(
    const float* __restrict__ x, const int* __restrict__ target,
    const float* __restrict__ normbuf, float* __restrict__ out) {
  extern __shared__ char smemraw[];
  _Float16* smB = (_Float16*)smemraw;           // 64 KB: 64 blocks x 1 KB
  int* t1s = (int*)(smemraw + 65536);           // 512 B

  const int p = blockIdx.x;
  const int b = blockIdx.y;

  // decode upper-triangular pair (i, j) from p (row-major triu, k=1)
  int i = 0, rem = p;
  while (rem >= (S_ - 1 - i)) { rem -= (S_ - 1 - i); ++i; }
  const int j = i + 1 + rem;

  const float* __restrict__ X0 = x + (size_t)(b * S_ + i) * W_ * H_;
  const float* __restrict__ X1 = x + (size_t)(b * S_ + j) * W_ * H_;
  const int tid = threadIdx.x;

  // ---- stage B = X1 (128x256 f32) into LDS as f16, fragment-native layout ----
  // fragment block (kk,nt) = 1 KB; lane chunk0 (K lo 8) packed in first 512 B,
  // chunk1 (K hi 8) in second 512 B -> ds_load_b128 spreads over all 64 banks.
  for (int g = tid; g < (W_ * H_) / 4; g += 256) {   // 8192 float4, coalesced
    const int r  = g >> 6;                // row 0..127
    const int kq = (g & 63) << 2;         // k, multiple of 4
    f4 v = ((const f4*)X1)[g];
    const int kk = kq >> 5;
    const int ko = kq & 31;
    const int hi = (ko >> 3) & 1;
    const int hf = ko >> 4;
    const int off = ko & 7;               // 0 or 4
    const int n = r & 15, nt = r >> 4;
    const int l = hi * 16 + n;
    _Float16* dst = smB + (kk * 8 + nt) * 512 + hf * 256 + l * 8 + off;
    *(h4*)dst = (h4){(_Float16)v.x, (_Float16)v.y, (_Float16)v.z, (_Float16)v.w};
  }
  if (tid < W_) t1s[tid] = target[(size_t)(b * S_ + j) * W_ + tid];
  __syncthreads();

  // ---- WMMA GEMM: wave owns 16-row strip, 8 N-tiles, K = 8 steps of 32 ----
  const float scale = 5.0f / normbuf[0];  // s / ||x||^2
  const int wave = tid >> 5, lane = tid & 31;
  const int m = lane & 15, hi = lane >> 4;
  const int rowBase = wave * 16;
  const float* __restrict__ Arow = X0 + (size_t)(rowBase + m) * H_;
  const _Float16* __restrict__ bbase = smB + lane * 8;

  v8f acc[8];
#pragma unroll
  for (int nt = 0; nt < 8; ++nt) acc[nt] = (v8f){};

  // prologue: A fragment ping-pong buffers; B fragments at distance-2
  v16h a0buf, a1buf = {};
  {
    f4 t0 = *(const f4*)(Arow + hi * 8);
    f4 t1 = *(const f4*)(Arow + hi * 8 + 4);
    f4 t2 = *(const f4*)(Arow + 16 + hi * 8);
    f4 t3 = *(const f4*)(Arow + 16 + hi * 8 + 4);
    a0buf = cvt16(t0, t1, t2, t3);
  }
  v8h c0 = *(const v8h*)(bbase);          // current  (step 0)
  v8h c1 = *(const v8h*)(bbase + 256);
  v8h d0 = *(const v8h*)(bbase + 512);    // next     (step 1)
  v8h d1 = *(const v8h*)(bbase + 768);
  f4 ra0 = {}, ra1 = {}, ra2 = {}, ra3 = {};

  // flattened 64-step pipeline: step = kk*8 + nt
#pragma unroll
  for (int step = 0; step < 64; ++step) {
    const int kk = step >> 3;
    const int nt = step & 7;

    // prefetch A fragment for kk+1 eight steps ahead (global, long latency)
    if (nt == 0 && kk < 7) {
      const float* ap = Arow + (kk + 1) * 32 + hi * 8;
      ra0 = *(const f4*)(ap);
      ra1 = *(const f4*)(ap + 4);
      ra2 = *(const f4*)(ap + 16);
      ra3 = *(const f4*)(ap + 20);
    }

    // prefetch B fragment for step+2 (LDS, distance-2 pipeline)
    v8h e0 = d0, e1 = d1;
    if (step + 2 < 64) {
      const _Float16* bp = bbase + (size_t)(step + 2) * 512;
      e0 = *(const v8h*)(bp);
      e1 = *(const v8h*)(bp + 256);
    }

    v16h bf = __builtin_shufflevector(c0, c1, 0, 1, 2, 3, 4, 5, 6, 7, 8, 9,
                                      10, 11, 12, 13, 14, 15);
    acc[nt] = __builtin_amdgcn_wmma_f32_16x16x32_f16(
        false, (kk & 1) ? a1buf : a0buf, false, bf, (short)0, acc[nt], false,
        false);

    c0 = d0; c1 = d1; d0 = e0; d1 = e1;

    // convert prefetched A into the *other* buffer before the last WMMA of
    // this kk: live ranges overlap -> distinct registers -> no WMMA->VALU
    // WAR hazard NOPs at the kk boundary.
    if (nt == 6 && kk < 7) {
      if ((kk + 1) & 1) a1buf = cvt16(ra0, ra1, ra2, ra3);
      else              a0buf = cvt16(ra0, ra1, ra2, ra3);
    }
  }

  // ---- epilogue: scale + store (C/D layout: row = hi*8+r, col = lane&15) ----
  float* __restrict__ Crow = out + (size_t)(b * P_ + p) * W_ * W_;
#pragma unroll
  for (int nt = 0; nt < 8; ++nt) {
    const int col = nt * 16 + m;
#pragma unroll
    for (int r = 0; r < 8; ++r) {
      const int row = rowBase + hi * 8 + r;
      Crow[(size_t)row * W_ + col] = acc[nt][r] * scale;
    }
  }

  // ---- fused target argmax: first v with t1[v]==t0[w], else 0 ----
  if (tid < W_) {
    const int ti = target[(size_t)(b * S_ + i) * W_ + tid];
    int best = 0;
    for (int v = W_ - 1; v >= 0; --v)
      if (t1s[v] == ti) best = v;
    out[(size_t)B_ * P_ * W_ * W_ + (size_t)(b * P_ + p) * W_ + tid] =
        (float)best;
  }
}

// ---------------- launcher ----------------
extern "C" void kernel_launch(void* const* d_in, const int* in_sizes, int n_in,
                              void* d_out, int out_size, void* d_ws,
                              size_t ws_size, hipStream_t stream) {
  (void)in_sizes; (void)n_in; (void)out_size; (void)ws_size;
  const float* x      = (const float*)d_in[0];
  const int*   target = (const int*)d_in[1];
  float* out = (float*)d_out;
  float* ws  = (float*)d_ws;

  float* normbuf  = ws;        // ws[0] = ||x||^2
  float* partials = ws + 16;   // NPART floats

  const int n4 = (B_ * S_ * W_ * H_) / 4;
  arcface_sumsq_partial<<<NPART, 256, 0, stream>>>(x, partials, n4);
  arcface_sumsq_final<<<1, 256, 0, stream>>>(partials, normbuf, NPART);

  dim3 grid(P_, B_);
  const size_t shmem = 65536 + W_ * sizeof(int);
  arcface_scores_kernel<<<grid, 256, shmem, stream>>>(x, target, normbuf, out);
}
